// Block_40913858461929
// MI455X (gfx1250) — matrix operand order repaired and checked by reading, not compiled, not run on hardware
//
#include <hip/hip_runtime.h>

// ---------------------------------------------------------------------------
// MI455X (gfx1250) Mamba block.
// GEMMs are matrix-core bound (~192 GFLOP vs ~300MB of traffic => compute
// bound at 23.3 TB/s), so all large matmuls go through v_wmma_f32_16x16x32_bf16
// (f32 accumulate). A-tiles are staged global->LDS with CDNA5 async-to-LDS
// DMA (global_load_async_to_lds_b128 + s_wait_asynccnt) with double-buffered
// LDS so DMA overlaps WMMA. The selective scan is parallelized over the state
// dimension: 16 lanes per (batch,channel), 1 exp/step/lane, and the state
// contraction is a 4-step __shfl_xor butterfly (wave32 cross-lane ops).
// ---------------------------------------------------------------------------

typedef __attribute__((ext_vector_type(16))) __bf16 bf16x16;
typedef __attribute__((ext_vector_type(8)))  __bf16 bf16x8;
typedef __attribute__((ext_vector_type(8)))  float  f32x8;

#define DEV static __device__ __forceinline__

DEV __bf16 f2bf(float f) {
    unsigned u = __builtin_bit_cast(unsigned, f);
    unsigned r = u + 0x7FFFu + ((u >> 16) & 1u);   // round-to-nearest-even
    unsigned short h = (unsigned short)(r >> 16);
    return __builtin_bit_cast(__bf16, h);
}
DEV float bf2f(__bf16 b) {
    unsigned short h = __builtin_bit_cast(unsigned short, b);
    return __builtin_bit_cast(float, ((unsigned)h) << 16);
}
DEV float silu_f(float v) { return v / (1.f + __expf(-v)); }

// CDNA5 async global->LDS DMA (ASYNCcnt-tracked); 16B per lane.
DEV void async_a2lds_b128(unsigned lds_byte_addr, const __bf16* gptr) {
    asm volatile("global_load_async_to_lds_b128 %0, %1, off"
                 :: "v"(lds_byte_addr), "v"(gptr)
                 : "memory");
}
DEV void wait_asynccnt0() {
    asm volatile("s_wait_asynccnt 0x0" ::: "memory");
}
DEV unsigned lds_addr32(const void* p) {
    // generic pointer to LDS: low 32 bits are the LDS byte address
    return (unsigned)(unsigned long long)p;
}

// ---------------------------------------------------------------------------
// f32 -> bf16 conversion (weights)
// ---------------------------------------------------------------------------
__global__ void cvt_bf16_kernel(const float* __restrict__ in,
                                __bf16* __restrict__ out, int n) {
    int i = blockIdx.x * blockDim.x + threadIdx.x;
    if (i < n) out[i] = f2bf(in[i]);
}

// ---------------------------------------------------------------------------
// Fused LayerNorm + RMSNorm (one row of 1024 per 256-thread block).
// ---------------------------------------------------------------------------
__global__ __launch_bounds__(256)
void ln_rms_kernel(const float* __restrict__ x,
                   const float* __restrict__ g, const float* __restrict__ b,
                   const float* __restrict__ rw,
                   float* __restrict__ hln, __bf16* __restrict__ xn) {
    const int D = 1024;
    const int row = blockIdx.x;
    const int tid = threadIdx.x;
    const float* xr = x + (size_t)row * D;
    __shared__ float rA[256], rB[256];

    float v[4]; float s = 0.f, ss = 0.f;
#pragma unroll
    for (int i = 0; i < 4; ++i) { v[i] = xr[tid + i * 256]; s += v[i]; ss += v[i] * v[i]; }
    rA[tid] = s; rB[tid] = ss; __syncthreads();
    for (int off = 128; off > 0; off >>= 1) {
        if (tid < off) { rA[tid] += rA[tid + off]; rB[tid] += rB[tid + off]; }
        __syncthreads();
    }
    float mu = rA[0] / D;
    float var = rB[0] / D - mu * mu;
    float rs = rsqrtf(var + 1e-5f);

    float h[4]; float hs = 0.f;
#pragma unroll
    for (int i = 0; i < 4; ++i) {
        int idx = tid + i * 256;
        h[i] = (v[i] - mu) * rs * g[idx] + b[idx];
        hln[(size_t)row * D + idx] = h[i];
        hs += h[i] * h[i];
    }
    __syncthreads();
    rA[tid] = hs; __syncthreads();
    for (int off = 128; off > 0; off >>= 1) {
        if (tid < off) rA[tid] += rA[tid + off];
        __syncthreads();
    }
    float r2 = rsqrtf(rA[0] / D + 1e-5f);
#pragma unroll
    for (int i = 0; i < 4; ++i) {
        int idx = tid + i * 256;
        xn[(size_t)row * D + idx] = f2bf(h[i] * r2 * rw[idx]);
    }
}

// ---------------------------------------------------------------------------
// Residual combine + LayerNorm2: x2 = x + h_ln + mix; h2 = ln2(x2) (bf16).
// ---------------------------------------------------------------------------
__global__ __launch_bounds__(256)
void resid_ln2_kernel(const float* __restrict__ x, const float* __restrict__ hln,
                      const float* __restrict__ mix,
                      const float* __restrict__ g, const float* __restrict__ b,
                      float* __restrict__ x2, __bf16* __restrict__ h2) {
    const int D = 1024;
    const int row = blockIdx.x;
    const int tid = threadIdx.x;
    const size_t base = (size_t)row * D;
    __shared__ float rA[256], rB[256];

    float v[4]; float s = 0.f, ss = 0.f;
#pragma unroll
    for (int i = 0; i < 4; ++i) {
        int idx = tid + i * 256;
        v[i] = x[base + idx] + hln[base + idx] + mix[base + idx];
        x2[base + idx] = v[i];
        s += v[i]; ss += v[i] * v[i];
    }
    rA[tid] = s; rB[tid] = ss; __syncthreads();
    for (int off = 128; off > 0; off >>= 1) {
        if (tid < off) { rA[tid] += rA[tid + off]; rB[tid] += rB[tid + off]; }
        __syncthreads();
    }
    float mu = rA[0] / D;
    float var = rB[0] / D - mu * mu;
    float rs = rsqrtf(var + 1e-5f);
#pragma unroll
    for (int i = 0; i < 4; ++i) {
        int idx = tid + i * 256;
        h2[base + idx] = f2bf((v[i] - mu) * rs * g[idx] + b[idx]);
    }
}

// ---------------------------------------------------------------------------
// Depthwise causal conv (K=4) + bias + SiLU over xz[:, :, :1024].
// ---------------------------------------------------------------------------
__global__ __launch_bounds__(256)
void conv_silu_kernel(const float* __restrict__ xz, const float* __restrict__ cw,
                      const float* __restrict__ cb,
                      float* __restrict__ xc, __bf16* __restrict__ xcb) {
    int idx = blockIdx.x * 256 + threadIdx.x;   // exactly 8192*1024 threads
    int e  = idx & 1023;
    int bt = idx >> 10;
    int t  = bt & 2047;
    int bb = bt >> 11;
    float s = cb[e];
#pragma unroll
    for (int j = 0; j < 4; ++j) {
        int tt = t - 3 + j;
        if (tt >= 0) s += cw[e * 4 + j] * xz[(size_t)(bb * 2048 + tt) * 2048 + e];
    }
    float o = silu_f(s);
    xc[idx]  = o;
    xcb[idx] = f2bf(o);
}

// ---------------------------------------------------------------------------
// Selective scan, state-parallel: 16 lanes cooperate on one (batch, channel),
// each lane owns one of the 16 states. Per timestep: 1 exp + FMAs per lane,
// then a 4-step __shfl_xor butterfly (masks 1/2/4/8 stay inside the 16-lane
// group) reduces sum_n h[n]*C[n]; lane n==0 applies D*x + silu(z) gate and
// stores. 65536 lanes = 2048 wave32s (vs 128 in the naive layout), and the
// per-step critical path drops from 16 serial exps to 1 exp + 4 shuffles.
// ---------------------------------------------------------------------------
__global__ __launch_bounds__(256)
void scan_kernel(const float* __restrict__ delta, const float* __restrict__ xc,
                 const __bf16* __restrict__ dbc, const float* __restrict__ xz,
                 const float* __restrict__ A_log, const float* __restrict__ Dp,
                 __bf16* __restrict__ y) {
    const int ch = blockIdx.x * 16 + (threadIdx.x >> 4);   // 0..4095
    const int n  = threadIdx.x & 15;                       // state index
    const int e  = ch & 1023;
    const int bb = ch >> 10;

    const float a  = -__expf(A_log[e * 16 + n]);
    const float Dv = Dp[e];
    float h = 0.f;

    for (int t = 0; t < 2048; ++t) {
        const size_t row = (size_t)bb * 2048 + t;
        const float d  = delta[row * 1024 + e];            // uniform in group
        const float xv = xc[row * 1024 + e];               // uniform in group
        const float Bn = bf2f(dbc[row * 96 + 64 + n]);     // coalesced
        const float Cn = bf2f(dbc[row * 96 + 80 + n]);     // coalesced
        h = __expf(d * a) * h + (d * xv) * Bn;
        float p = h * Cn;
        p += __shfl_xor(p, 1);
        p += __shfl_xor(p, 2);
        p += __shfl_xor(p, 4);
        p += __shfl_xor(p, 8);
        if (n == 0) {
            const float zv = xz[row * 2048 + 1024 + e];
            const float acc = p + Dv * xv;
            y[row * 1024 + e] = f2bf(acc * silu_f(zv));
        }
    }
}

// ---------------------------------------------------------------------------
// bf16 WMMA GEMM: C[M,N] = A[M,K] @ B[K,N] (+epilogue). Block tile 128x128,
// K-step 32, 8 waves: wave (wm 0..3, wn 0..1) computes a 32x64 patch as a
// 2x4 grid of v_wmma_f32_16x16x32_bf16 tiles. Double-buffered LDS:
//   A tile: async global->LDS DMA (ASYNCcnt), contiguous 16B/lane.
//   B tile: global b128 loads into regs, transposed scatter to LDS (so the
//           B fragment is a contiguous 32B LDS read per lane).
// Per K-step: scatter B(cur) -> wait asynccnt 0 -> barrier -> issue async
// A(next) + load B(next) regs -> 8 WMMAs on (cur). The barrier guarantees the
// 'next' buffer's previous readers are done before the DMA lands.
// NG templates away the N-bound guard (only the N=96 x_proj GEMM needs it).
// MODE: 0 f32 | 1 bf16 | 2 softplus(acc+bias)->f32 | 3 relu(acc+bias)->bf16
//       4 acc+bias+res->f32
// ---------------------------------------------------------------------------
template <int MODE, bool NG>
__global__ __launch_bounds__(256)
void gemm_bf16(const __bf16* __restrict__ A, int lda,
               const __bf16* __restrict__ B, int ldb,
               void* __restrict__ out, int ldo,
               const float* __restrict__ bias,
               const float* __restrict__ res,
               int N, int K) {
    __shared__ __bf16 sA[2][128][32];   // [buf][m][k]
    __shared__ __bf16 sB[2][128][32];   // [buf][n][k] (transposed)

    const int tid  = threadIdx.x;
    const int lane = tid & 31;
    const int w    = tid >> 5;
    const int wm   = w >> 1;          // 0..3
    const int wn   = w & 1;           // 0..1
    const int m0   = blockIdx.y * 128;
    const int n0   = blockIdx.x * 128;
    const int hf   = lane >> 4;       // lane half (0/1)
    const int l15  = lane & 15;

    const f32x8 zacc = {0.f, 0.f, 0.f, 0.f, 0.f, 0.f, 0.f, 0.f};
    f32x8 c[2][4];
#pragma unroll
    for (int mi = 0; mi < 2; ++mi)
#pragma unroll
        for (int ni = 0; ni < 4; ++ni) c[mi][ni] = zacc;

    // per-thread tile-fill coordinates (256 threads x 16 elems = 4096 = tile)
    const int ar = tid >> 1;              // A row 0..127
    const int ak = (tid & 1) << 4;        // A k offset 0/16
    const int bk = tid >> 3;              // B k 0..31
    const int bn = (tid & 7) << 4;        // B n offset 0..112
    const bool bok = (!NG) || (n0 + bn) < N;   // Ns are multiples of 16

    const __bf16* aG = A + (size_t)(m0 + ar) * lda + ak;       // += 32 per step
    const __bf16* bG = B + (size_t)bk * ldb + n0 + bn;         // += 32*ldb

    const int ksteps = K >> 5;

    // ---- prologue: stage k-step 0 ----
    async_a2lds_b128(lds_addr32(&sA[0][ar][ak]), aG);
    async_a2lds_b128(lds_addr32(&sA[0][ar][ak]) + 16u, aG + 8);
    uint4 bv0 = make_uint4(0, 0, 0, 0), bv1 = make_uint4(0, 0, 0, 0);
    if (bok) {
        const uint4* bp = (const uint4*)bG;
        bv0 = bp[0]; bv1 = bp[1];
    }

    for (int kb = 0; kb < ksteps; ++kb) {
        const int cur = kb & 1;
        const int nxt = cur ^ 1;

        // (a) transposed scatter of this step's B registers into sB[cur]
        {
            __bf16 tmp[16];
            *(uint4*)(&tmp[0]) = bv0;
            *(uint4*)(&tmp[8]) = bv1;
#pragma unroll
            for (int j = 0; j < 16; ++j) sB[cur][bn + j][bk] = tmp[j];
        }
        // (b) our async A(cur) transfers complete
        wait_asynccnt0();
        // (c) all waves: A(cur) landed, B(cur) stores visible, prev readers done
        __syncthreads();
        // (d,e) stage k-step kb+1 into the other buffer (overlaps the WMMAs)
        if (kb + 1 < ksteps) {
            const __bf16* aN = aG + (size_t)(kb + 1) * 32;
            unsigned l0 = lds_addr32(&sA[nxt][ar][ak]);
            async_a2lds_b128(l0, aN);
            async_a2lds_b128(l0 + 16u, aN + 8);
            if (bok) {
                const uint4* bp = (const uint4*)(bG + (size_t)(kb + 1) * 32 * ldb);
                bv0 = bp[0]; bv1 = bp[1];
            }
        }

        // (f) fragments + 8 WMMAs on buffer 'cur'
        // A frag: lane(m=l15, half=hf): elems i = K[hf*16+i]
        bf16x16 af[2];
#pragma unroll
        for (int mi = 0; mi < 2; ++mi) {
            const __bf16* pa = &sA[cur][wm * 32 + mi * 16 + l15][0];
            bf16x8 lo = *(const bf16x8*)(pa + hf * 16);
            bf16x8 hi = *(const bf16x8*)(pa + hf * 16 + 8);
            af[mi] = __builtin_shufflevector(lo, hi, 0, 1, 2, 3, 4, 5, 6, 7,
                                             8, 9, 10, 11, 12, 13, 14, 15);
        }
        // B frag: lane(n=l15, half=hf): elems i = B[hf*16+i][n] (contiguous in sB^T)
        bf16x16 bfr[4];
#pragma unroll
        for (int ni = 0; ni < 4; ++ni)
            bfr[ni] = *(const bf16x16*)(&sB[cur][wn * 64 + ni * 16 + l15][hf * 16]);

#pragma unroll
        for (int mi = 0; mi < 2; ++mi)
#pragma unroll
            for (int ni = 0; ni < 4; ++ni)
                c[mi][ni] = __builtin_amdgcn_wmma_f32_16x16x32_bf16(
                    false, af[mi], false, bfr[ni], (short)0, c[mi][ni], false, false);
    }

    // epilogue: D layout => VGPR r, lane: m = r + hf*8, n = l15
#pragma unroll
    for (int mi = 0; mi < 2; ++mi) {
#pragma unroll
        for (int ni = 0; ni < 4; ++ni) {
            const int gn = n0 + wn * 64 + ni * 16 + l15;
            if (NG && gn >= N) continue;
            const int gmBase = m0 + wm * 32 + mi * 16 + hf * 8;
#pragma unroll
            for (int r = 0; r < 8; ++r) {
                const int gm = gmBase + r;
                float v = c[mi][ni][r];
                const size_t o = (size_t)gm * ldo + gn;
                if (MODE == 0) {
                    ((float*)out)[o] = v;
                } else if (MODE == 1) {
                    ((__bf16*)out)[o] = f2bf(v);
                } else if (MODE == 2) {
                    v += bias[gn];
                    ((float*)out)[o] = (v > 20.f) ? v : log1pf(__expf(v));
                } else if (MODE == 3) {
                    v += bias[gn];
                    ((__bf16*)out)[o] = f2bf(fmaxf(v, 0.f));
                } else {
                    v += bias[gn] + res[o];
                    ((float*)out)[o] = v;
                }
            }
        }
    }
}

// ---------------------------------------------------------------------------
// Host orchestration
// ---------------------------------------------------------------------------
extern "C" void kernel_launch(void* const* d_in, const int* in_sizes, int n_in,
                              void* d_out, int out_size, void* d_ws, size_t ws_size,
                              hipStream_t stream) {
    (void)in_sizes; (void)n_in; (void)out_size; (void)ws_size;

    const float* x        = (const float*)d_in[0];
    const float* ln1_g    = (const float*)d_in[1];
    const float* ln1_b    = (const float*)d_in[2];
    const float* ln2_g    = (const float*)d_in[3];
    const float* ln2_b    = (const float*)d_in[4];
    const float* rms_w    = (const float*)d_in[5];
    const float* in_proj  = (const float*)d_in[6];
    const float* conv_w   = (const float*)d_in[7];
    const float* conv_b   = (const float*)d_in[8];
    const float* x_proj   = (const float*)d_in[9];
    const float* dt_proj  = (const float*)d_in[10];
    const float* dt_b     = (const float*)d_in[11];
    const float* A_log    = (const float*)d_in[12];
    const float* D_param  = (const float*)d_in[13];
    const float* out_proj = (const float*)d_in[14];
    const float* ffn_w1   = (const float*)d_in[15];
    const float* ffn_b1   = (const float*)d_in[16];
    const float* ffn_w2   = (const float*)d_in[17];
    const float* ffn_b2   = (const float*)d_in[18];

    const int M = 4 * 2048;   // 8192 rows

    // Workspace layout (bytes, lifetime-based reuse; ~226 MB)
    char* ws = (char*)d_ws;
    __bf16* W_INPROJ  = (__bf16*)(ws + 0);              // 1024*2048 bf16
    __bf16* W_XPROJ   = (__bf16*)(ws + 4194304);        // 1024*96
    __bf16* W_DTPROJ  = (__bf16*)(ws + 4390912);        // 64*1024
    __bf16* W_OUTPROJ = (__bf16*)(ws + 4521984);        // 1024*1024
    __bf16* W_FFN1    = (__bf16*)(ws + 6619136);        // 1024*4096
    __bf16* W_FFN2    = (__bf16*)(ws + 15007744);       // 4096*1024
    float*  H_LN      = (float*)(ws + 23396352);        // M*1024 f32
    __bf16* XN        = (__bf16*)(ws + 56950784);       // M*1024 bf16 (reused: H2)
    float*  XZ        = (float*)(ws + 73728000);        // M*2048 f32 (reused: FFN act)
    float*  XC        = (float*)(ws + 140836864);       // M*1024 f32 (reused: X2)
    __bf16* XCB       = (__bf16*)(ws + 174391296);      // M*1024 bf16 (reused: Y)
    __bf16* DBC       = (__bf16*)(ws + 191168512);      // M*96 bf16
    float*  DELTA     = (float*)(ws + 192741376);       // M*1024 f32 (reused: MIX)
    __bf16* H2   = XN;
    __bf16* ACT  = (__bf16*)XZ;
    float*  X2   = XC;
    __bf16* Y    = XCB;
    float*  MIX  = DELTA;

    // 1) weights -> bf16
    cvt_bf16_kernel<<<(1024 * 2048 + 255) / 256, 256, 0, stream>>>(in_proj, W_INPROJ, 1024 * 2048);
    cvt_bf16_kernel<<<(1024 * 96 + 255) / 256, 256, 0, stream>>>(x_proj, W_XPROJ, 1024 * 96);
    cvt_bf16_kernel<<<(64 * 1024 + 255) / 256, 256, 0, stream>>>(dt_proj, W_DTPROJ, 64 * 1024);
    cvt_bf16_kernel<<<(1024 * 1024 + 255) / 256, 256, 0, stream>>>(out_proj, W_OUTPROJ, 1024 * 1024);
    cvt_bf16_kernel<<<(1024 * 4096 + 255) / 256, 256, 0, stream>>>(ffn_w1, W_FFN1, 1024 * 4096);
    cvt_bf16_kernel<<<(4096 * 1024 + 255) / 256, 256, 0, stream>>>(ffn_w2, W_FFN2, 4096 * 1024);

    // 2) LN1 + RMSNorm
    ln_rms_kernel<<<M, 256, 0, stream>>>(x, ln1_g, ln1_b, rms_w, H_LN, XN);

    // 3) xz = xn @ in_proj  (8192 x 1024 x 2048) -> f32
    gemm_bf16<0, false><<<dim3(2048 / 128, M / 128), 256, 0, stream>>>(
        XN, 1024, W_INPROJ, 2048, (void*)XZ, 2048, nullptr, nullptr, 2048, 1024);

    // 4) depthwise conv + silu -> xc (f32 + bf16)
    conv_silu_kernel<<<(M * 1024) / 256, 256, 0, stream>>>(XZ, conv_w, conv_b, XC, XCB);

    // 5) dbc = xc @ x_proj  (8192 x 1024 x 96) -> bf16 (N guard on)
    gemm_bf16<1, true><<<dim3(1, M / 128), 256, 0, stream>>>(
        XCB, 1024, W_XPROJ, 96, (void*)DBC, 96, nullptr, nullptr, 96, 1024);

    // 6) delta = softplus(dbc[:, :64] @ dt_proj + dt_b)  (8192 x 64 x 1024) -> f32
    gemm_bf16<2, false><<<dim3(1024 / 128, M / 128), 256, 0, stream>>>(
        DBC, 96, W_DTPROJ, 1024, (void*)DELTA, 1024, dt_b, nullptr, 1024, 64);

    // 7) selective scan, state-parallel -> y (bf16, reuses XCB)
    scan_kernel<<<256, 256, 0, stream>>>(DELTA, XC, DBC, XZ, A_log, D_param, Y);

    // 8) mix = y @ out_proj  (8192 x 1024 x 1024) -> f32 (reuses DELTA)
    gemm_bf16<0, false><<<dim3(1024 / 128, M / 128), 256, 0, stream>>>(
        Y, 1024, W_OUTPROJ, 1024, (void*)MIX, 1024, nullptr, nullptr, 1024, 1024);

    // 9) x2 = x + h_ln + mix ; h2 = ln2(x2)
    resid_ln2_kernel<<<M, 256, 0, stream>>>(x, H_LN, MIX, ln2_g, ln2_b, X2, H2);

    // 10) act = relu(h2 @ ffn_w1 + b1)  (8192 x 1024 x 4096) -> bf16 (reuses XZ)
    gemm_bf16<3, false><<<dim3(4096 / 128, M / 128), 256, 0, stream>>>(
        H2, 1024, W_FFN1, 4096, (void*)ACT, 4096, ffn_b1, nullptr, 4096, 1024);

    // 11) out = act @ ffn_w2 + b2 + x2  (8192 x 4096 x 1024) -> f32 to d_out
    gemm_bf16<4, false><<<dim3(1024 / 128, M / 128), 256, 0, stream>>>(
        ACT, 4096, W_FFN2, 1024, d_out, 1024, ffn_b2, X2, 1024, 4096);
}
